// GATBackbone_89550068122074
// MI455X (gfx1250) — compile-verified, matrix operand order checked by reading, not run on hardware
//
#include <hip/hip_runtime.h>
#include <hip/hip_bf16.h>

// ---------------------------------------------------------------------------
// GAT backbone on MI455X (gfx1250).
//  - GEMMs (x@W): bf16 WMMA with operands PRE-PACKED into the exact CDNA5
//    fragment layout so each lane's fragment is one 32B contiguous load.
//  - Edge softmax/aggregation: flat atomic passes (memory-bound, L2-resident).
// ---------------------------------------------------------------------------

#define N_NODES  50000
#define N_EDGES  800000
#define N_GRAPHS 512
#define NEG_SLOPE 0.2f

typedef __bf16 bf16_t;
typedef __attribute__((ext_vector_type(16))) __bf16 v16bf;
typedef __attribute__((ext_vector_type(8)))  float  v8f;

// ---------------- utility kernels ----------------
__global__ void zero_f32(float* p, int n) {
  int i = blockIdx.x * blockDim.x + threadIdx.x;
  if (i < n) p[i] = 0.f;
}
__global__ void fill_u32(unsigned* p, unsigned v, int n) {
  int i = blockIdx.x * blockDim.x + threadIdx.x;
  if (i < n) p[i] = v;
}

// ---------------- fragment packing (f32 -> bf16, WMMA layout) ----------------
// CDNA5 16-bit A fragment for a 16x32 (MxK) tile (ISA 7.12.2):
//   lane L: row m = L&15 (+ tile base); K values = 8*(L>>4) + {0..7} and
//   16 + 8*(L>>4) + {0..7}; element j of the 16-wide fragment maps to
//   k = 8*(L>>4) + (j<8 ? j : j+8).
// Packed layout: out[((mt*nKT + kt)*32 + lane)*16 + j]  (32B per lane, coalesced)
__global__ void pack_a_frag(const float* __restrict__ in, bf16_t* __restrict__ out,
                            int M, int K) {
  int t = blockIdx.x * blockDim.x + threadIdx.x;
  if (t >= M * K) return;
  int j    = t & 15;
  int lane = (t >> 4) & 31;
  int tile = t >> 9;                 // mt*nKT + kt
  int nKT  = K >> 5;
  int mt = tile / nKT, kt = tile % nKT;
  int m = (mt << 4) + (lane & 15);
  int k = (kt << 5) + 8 * (lane >> 4) + (j < 8 ? j : j + 8);
  out[t] = (bf16_t)in[(size_t)m * K + k];
}
// B [K,Nc] row-major -> fragment-major: out[((nt*nKT + kt)*32 + lane)*16 + j]
// lane L holds column n = nt*16 + (L&15), same K packing as A.
__global__ void pack_b_frag(const float* __restrict__ in, bf16_t* __restrict__ out,
                            int K, int Nc) {
  int t = blockIdx.x * blockDim.x + threadIdx.x;
  if (t >= K * Nc) return;
  int j    = t & 15;
  int lane = (t >> 4) & 31;
  int tile = t >> 9;                 // nt*nKT + kt
  int nKT  = K >> 5;
  int nt = tile / nKT, kt = tile % nKT;
  int n = (nt << 4) + (lane & 15);
  int k = (kt << 5) + 8 * (lane >> 4) + (j < 8 ? j : j + 8);
  out[t] = (bf16_t)in[(size_t)k * Nc + n];
}

// ---------------- WMMA GEMM on packed fragments ----------------
// One wave per 16-row strip; computes all NT 16-col tiles of that strip,
// reusing the A fragment across NT back-to-back WMMAs per k-step.
template <int NT>
__global__ void __launch_bounds__(256)
wmma_gemm_packed(const bf16_t* __restrict__ Ap, const bf16_t* __restrict__ Bp,
                 float* __restrict__ C, int M, int K, int Nc) {
  const int wave = blockIdx.x * (blockDim.x >> 5) + (threadIdx.x >> 5);
  const int nMT  = M >> 4;
  if (wave >= nMT) return;
  const int lane = threadIdx.x & 31;
  const int half = lane >> 4;
  const int l15  = lane & 15;
  const int nKT  = K >> 5;

  v8f acc[NT];
#pragma unroll
  for (int i = 0; i < NT; ++i) acc[i] = (v8f){};

  const v16bf* a_base = (const v16bf*)Ap + (size_t)wave * nKT * 32 + lane;
  const v16bf* b_base = (const v16bf*)Bp + lane;

  for (int kt = 0; kt < nKT; ++kt) {
    v16bf a = a_base[(size_t)kt * 32];               // one 32B contiguous load
#pragma unroll
    for (int nt = 0; nt < NT; ++nt) {
      v16bf b = b_base[((size_t)nt * nKT + kt) * 32];
      acc[nt] = __builtin_amdgcn_wmma_f32_16x16x32_bf16(
          /*neg_a=*/false, a, /*neg_b=*/false, b,
          /*c_mod=*/(short)0, acc[nt], /*reuse_a=*/false, /*reuse_b=*/false);
    }
  }
  // C/D layout: VGPR i -> M = i + 8*(lane>=16), N = lane&15
#pragma unroll
  for (int nt = 0; nt < NT; ++nt)
#pragma unroll
    for (int i = 0; i < 8; ++i)
      C[(size_t)((wave << 4) + i + 8 * half) * Nc + (nt << 4) + l15] = acc[nt][i];
}

// ---------------- per-(node,head) attention logits ----------------
__global__ void alpha_kernel(const float* __restrict__ h, const float* __restrict__ a_s,
                             const float* __restrict__ a_d, float* __restrict__ asrc,
                             float* __restrict__ adst, int H, int C) {
  int t = blockIdx.x * blockDim.x + threadIdx.x;
  if (t >= N_NODES * H) return;
  int hh = t % H, n = t / H;
  const float* hp = h + (size_t)n * H * C + (size_t)hh * C;
  const float* sp = a_s + hh * C;
  const float* dp = a_d + hh * C;
  float s = 0.f, d = 0.f;
#pragma unroll 8
  for (int c = 0; c < C; ++c) { float v = hp[c]; s += v * sp[c]; d += v * dp[c]; }
  asrc[t] = s;
  adst[t] = d;
}

// ---------------- edge helpers ----------------
__device__ __forceinline__ void edge_sd(const int* __restrict__ ei, int e, int& s, int& d) {
  if (e < N_EDGES) { s = ei[e]; d = ei[N_EDGES + e]; }
  else             { s = e - N_EDGES; d = s; }          // self-loop
}
__device__ __forceinline__ float lrelu(float v) { return v > 0.f ? v : NEG_SLOPE * v; }
// order-preserving float<->uint encoding for atomic max
__device__ __forceinline__ unsigned enc_f32(float v) {
  unsigned u = __float_as_uint(v);
  return u ^ ((u >> 31) ? 0xFFFFFFFFu : 0x80000000u);
}
__device__ __forceinline__ float dec_u32(unsigned u) {
  u ^= ((u >> 31) ? 0x80000000u : 0xFFFFFFFFu);
  return __uint_as_float(u);
}

// pass 1: segment max of leaky_relu(alpha_s[src]+alpha_d[dst]) over dst
__global__ void edge_max_k(const int* __restrict__ ei, const float* __restrict__ asrc,
                           const float* __restrict__ adst, unsigned* __restrict__ m, int H) {
  int t = blockIdx.x * blockDim.x + threadIdx.x;
  int total = (N_EDGES + N_NODES) * H;
  if (t >= total) return;
  int hh = t % H, e = t / H;
  int s, d; edge_sd(ei, e, s, d);
  float v = lrelu(asrc[s * H + hh] + adst[d * H + hh]);
  atomicMax(m + d * H + hh, enc_f32(v));
}

// pass 2: denom = segment_sum(exp(e - m[dst]))
__global__ void edge_expsum_k(const int* __restrict__ ei, const float* __restrict__ asrc,
                              const float* __restrict__ adst, const unsigned* __restrict__ m,
                              float* __restrict__ denom, int H) {
  int t = blockIdx.x * blockDim.x + threadIdx.x;
  int total = (N_EDGES + N_NODES) * H;
  if (t >= total) return;
  int hh = t % H, e = t / H;
  int s, d; edge_sd(ei, e, s, d);
  float v  = lrelu(asrc[s * H + hh] + adst[d * H + hh]);
  float mv = dec_u32(m[d * H + hh]);
  atomicAdd(denom + d * H + hh, __expf(v - mv));
}

// pass 3: out[dst] += alpha * h[src]  (32 channels per (edge,head))
__global__ void edge_msg_k(const int* __restrict__ ei, const float* __restrict__ asrc,
                           const float* __restrict__ adst, const unsigned* __restrict__ m,
                           const float* __restrict__ denom, const float* __restrict__ h,
                           float* __restrict__ out, int H, int C) {
  int t = blockIdx.x * blockDim.x + threadIdx.x;
  int total = (N_EDGES + N_NODES) * H;
  if (t >= total) return;
  int hh = t % H, e = t / H;
  int s, d; edge_sd(ei, e, s, d);
  float v  = lrelu(asrc[s * H + hh] + adst[d * H + hh]);
  float mv = dec_u32(m[d * H + hh]);
  float alpha = __expf(v - mv) / (denom[d * H + hh] + 1e-16f);
  const float* hp = h   + (size_t)s * H * C + (size_t)hh * C;
  float*       op = out + (size_t)d * H * C + (size_t)hh * C;
#pragma unroll 8
  for (int c = 0; c < C; ++c) atomicAdd(op + c, hp[c] * alpha);
}

// bias + ELU (in place)
__global__ void bias_elu_k(float* __restrict__ x, const float* __restrict__ b,
                           int HC, int total) {
  int i = blockIdx.x * blockDim.x + threadIdx.x;
  if (i >= total) return;
  float v = x[i] + b[i % HC];
  x[i] = v > 0.f ? v : (__expf(v) - 1.f);
}

// ---------------- global mean pool ----------------
__global__ void pool_acc_k(const float* __restrict__ h, const int* __restrict__ batch,
                           float* __restrict__ gsum, float* __restrict__ gcnt) {
  int t = blockIdx.x * blockDim.x + threadIdx.x;
  if (t >= N_NODES * 32) return;
  int c = t & 31, n = t >> 5;
  int g = batch[n];
  atomicAdd(gsum + (size_t)g * 32 + c, h[(size_t)n * 32 + c]);
  if (c == 0) atomicAdd(gcnt + g, 1.f);
}
__global__ void pool_div_k(const float* __restrict__ gsum, const float* __restrict__ gcnt,
                           float* __restrict__ out) {
  int t = blockIdx.x * blockDim.x + threadIdx.x;
  if (t >= N_GRAPHS * 32) return;
  out[t] = gsum[t] / fmaxf(gcnt[t >> 5], 1.f);
}

// ---------------------------------------------------------------------------
extern "C" void kernel_launch(void* const* d_in, const int* in_sizes, int n_in,
                              void* d_out, int out_size, void* d_ws, size_t ws_size,
                              hipStream_t stream) {
  (void)in_sizes; (void)n_in; (void)out_size; (void)ws_size;

  const float* x    = (const float*)d_in[0];
  const int*   ei   = (const int*)d_in[1];
  const int*   bidx = (const int*)d_in[2];
  const float* Wl[3]  = {(const float*)d_in[3],  (const float*)d_in[7],  (const float*)d_in[11]};
  const float* asl[3] = {(const float*)d_in[4],  (const float*)d_in[8],  (const float*)d_in[12]};
  const float* adl[3] = {(const float*)d_in[5],  (const float*)d_in[9],  (const float*)d_in[13]};
  const float* bl[3]  = {(const float*)d_in[6],  (const float*)d_in[10], (const float*)d_in[14]};

  // workspace carve-up (~67 MB total)
  char* ws = (char*)d_ws;
  size_t off = 0;
  auto take = [&](size_t bytes) -> char* {
    char* p = ws + off;
    off = (off + bytes + 255) & ~(size_t)255;
    return p;
  };
  float*    hA    = (float*)take((size_t)N_NODES * 128 * sizeof(float));   // GEMM result h
  float*    hB    = (float*)take((size_t)N_NODES * 128 * sizeof(float));   // aggregated / next input
  bf16_t*   apack = (bf16_t*)take((size_t)N_NODES * 128 * sizeof(bf16_t)); // packed A fragments
  bf16_t*   bpack = (bf16_t*)take((size_t)128 * 128 * sizeof(bf16_t));     // packed B fragments
  float*    asrc  = (float*)take((size_t)N_NODES * 4 * sizeof(float));
  float*    adst  = (float*)take((size_t)N_NODES * 4 * sizeof(float));
  unsigned* mbuf  = (unsigned*)take((size_t)N_NODES * 4 * sizeof(unsigned));
  float*    denom = (float*)take((size_t)N_NODES * 4 * sizeof(float));
  float*    gsum  = (float*)take((size_t)N_GRAPHS * 32 * sizeof(float));
  float*    gcnt  = (float*)take((size_t)N_GRAPHS * sizeof(float));

  const int Hs[3]  = {4, 4, 1};
  const int C = 32;
  const int F = 128;                 // F_in is 128 for all three layers

  const float* cur = x;
  for (int l = 0; l < 3; ++l) {
    const int H    = Hs[l];
    const int HC   = H * C;
    const int nIn  = N_NODES * F;
    const int nW   = F * HC;
    const int nOut = N_NODES * HC;
    const int nNH  = N_NODES * H;
    const int totE = (N_EDGES + N_NODES) * H;

    // 1) pack operands into WMMA fragment layout (bf16). Frees hB for reuse.
    pack_a_frag<<<(nIn + 255) / 256, 256, 0, stream>>>(cur, apack, N_NODES, F);
    pack_b_frag<<<(nW + 255) / 256, 256, 0, stream>>>(Wl[l], bpack, F, HC);

    // 2) h = x @ W : one wave per 16-row strip, NT tiles wide
    {
      int nWaves = N_NODES / 16;     // 3125
      int blocks = (nWaves + 7) / 8; // 8 waves / block
      if (HC == 128)
        wmma_gemm_packed<8><<<blocks, 256, 0, stream>>>(apack, bpack, hA, N_NODES, F, HC);
      else
        wmma_gemm_packed<2><<<blocks, 256, 0, stream>>>(apack, bpack, hA, N_NODES, F, HC);
    }

    // 3) attention logits per (node, head)
    alpha_kernel<<<(nNH + 255) / 256, 256, 0, stream>>>(hA, asl[l], adl[l], asrc, adst, H, C);

    // 4) init segment buffers (0 lies below enc(-inf) in the uint max-encoding)
    fill_u32<<<(nNH + 255) / 256, 256, 0, stream>>>(mbuf, 0u, nNH);
    zero_f32<<<(nNH + 255) / 256, 256, 0, stream>>>(denom, nNH);
    zero_f32<<<(nOut + 255) / 256, 256, 0, stream>>>(hB, nOut);

    // 5) edge passes: segment-max, exp-sum, weighted scatter
    edge_max_k   <<<(totE + 255) / 256, 256, 0, stream>>>(ei, asrc, adst, mbuf, H);
    edge_expsum_k<<<(totE + 255) / 256, 256, 0, stream>>>(ei, asrc, adst, mbuf, denom, H);
    edge_msg_k   <<<(totE + 255) / 256, 256, 0, stream>>>(ei, asrc, adst, mbuf, denom, hA, hB, H, C);

    // 6) bias + ELU
    bias_elu_k<<<(nOut + 255) / 256, 256, 0, stream>>>(hB, bl[l], HC, nOut);

    cur = hB;   // next layer input
  }

  // global mean pool -> d_out [G, 32]
  zero_f32<<<(N_GRAPHS * 32 + 255) / 256, 256, 0, stream>>>(gsum, N_GRAPHS * 32);
  zero_f32<<<(N_GRAPHS + 255) / 256, 256, 0, stream>>>(gcnt, N_GRAPHS);
  pool_acc_k<<<(N_NODES * 32 + 255) / 256, 256, 0, stream>>>(hB, bidx, gsum, gcnt);
  pool_div_k<<<(N_GRAPHS * 32 + 255) / 256, 256, 0, stream>>>(gsum, gcnt, (float*)d_out);
}
